// QLSTMGen108_65481071401187
// MI455X (gfx1250) — matrix operand order repaired and checked by reading, not compiled, not run on hardware
//
#include <hip/hip_runtime.h>
#include <hip/hip_bf16.h>

// ---------------- problem constants (match reference) ----------------
#define TT   512
#define BB   2048
#define DD   64
#define HH   10
#define NJ   40          // 4 gates * H outputs per row
#define DH   74          // D + H, weight row stride
#define CLIPV 5.0f

typedef __attribute__((ext_vector_type(16))) _Float16 v16h;
typedef __attribute__((ext_vector_type(8)))  float    v8f;

__device__ __forceinline__ const float* gate_wptr(int g, const float* Wf, const float* Wi,
                                                  const float* Wu, const float* Wo) {
    return g == 0 ? Wf : (g == 1 ? Wi : (g == 2 ? Wu : Wo));
}
__device__ __forceinline__ const float* gate_bptr(int g, const float* bf, const float* bi,
                                                  const float* bu, const float* bo) {
    return g == 0 ? bf : (g == 1 ? bi : (g == 2 ? bu : bo));
}

// =====================================================================
// Kernel A: Zx[(t*B+b)][j] = bias_j + x[t][b][:] . W_g[j%10][0:64]
// One wave computes a 16(M-rows) x 40(N-outputs) tile via split-precision
// f16 WMMA with f32 accumulate: acc += Ah*Bh + Ah*Bl + Al*Bh.
// 128 threads/block, min-occupancy launch bounds: trade occupancy for
// registers so the 12 hoisted B fragments stay resident (no scratch spills);
// parallelism comes from 8192 waves, the kernel is HBM-bound anyway.
// =====================================================================
__global__ void __launch_bounds__(128, 1)
qlstm_pregemm_wmma(const float* __restrict__ x,
                   const float* __restrict__ Wf, const float* __restrict__ bf,
                   const float* __restrict__ Wi, const float* __restrict__ bi,
                   const float* __restrict__ Wu, const float* __restrict__ bu,
                   const float* __restrict__ Wo, const float* __restrict__ bo,
                   float* __restrict__ Zx) {
    const int lane  = threadIdx.x & 31;
    const int hi    = lane >> 4;      // which 16-lane half
    const int ln    = lane & 15;
    const int wavesPerBlock = blockDim.x >> 5;
    const int waveId   = blockIdx.x * wavesPerBlock + (threadIdx.x >> 5);
    const int numWaves = gridDim.x * wavesPerBlock;
    const int numTiles = (TT * BB) / 16;   // 65536 M-tiles

    // ---- hoisted B fragments (3 N-tiles x 2 K-steps, hi/lo) and bias ----
    // B layout (16-bit, 32x16): lane = N (per half); half i -> K = 32*ks + 16*hi + i
    v16h bh[3][2], bl[3][2];
    float biasv[3];
    #pragma unroll
    for (int n = 0; n < 3; ++n) {
        const int j = n * 16 + ln;         // output column 0..47
        float bias = 0.0f;
        v16h z;
        #pragma unroll
        for (int i = 0; i < 16; ++i) z[i] = (_Float16)0.0f;
        bh[n][0] = z; bh[n][1] = z; bl[n][0] = z; bl[n][1] = z;
        if (j < NJ) {
            const int g  = j / HH;
            const int jr = j % HH;
            bias = gate_bptr(g, bf, bi, bu, bo)[jr];
            const float* w = gate_wptr(g, Wf, Wi, Wu, Wo) + (size_t)jr * DH;
            #pragma unroll
            for (int ks = 0; ks < 2; ++ks) {
                const int kb = 32 * ks + 16 * hi;
                v16h th, tl;
                #pragma unroll
                for (int i = 0; i < 16; ++i) {
                    const float v = w[kb + i];
                    const _Float16 h = (_Float16)v;
                    th[i] = h;
                    tl[i] = (_Float16)(v - (float)h);
                }
                bh[n][ks] = th; bl[n][ks] = tl;
            }
        }
        biasv[n] = bias;
    }

    for (int tile = waveId; tile < numTiles; tile += numWaves) {
        const size_t rowBase = (size_t)tile * 16;

        v8f acc[3];
        #pragma unroll
        for (int n = 0; n < 3; ++n) {
            v8f a;
            #pragma unroll
            for (int e = 0; e < 8; ++e) a[e] = biasv[n];
            acc[n] = a;
        }

        // A layout (16-bit, 16x32): M = ln; half i -> K = 32*ks + 16*hi + i
        const float* xrow = x + (rowBase + (size_t)ln) * DD;
        #pragma unroll
        for (int ks = 0; ks < 2; ++ks) {
            const int kb1 = 32 * ks + 16 * hi;
            const int kb2 = kb1 + 8;
            v16h ah, al;
            #pragma unroll
            for (int i = 0; i < 8; ++i) {
                const float v = xrow[kb1 + i];
                const _Float16 h = (_Float16)v;
                ah[i] = h;
                al[i] = (_Float16)(v - (float)h);
            }
            #pragma unroll
            for (int i = 0; i < 8; ++i) {
                const float v = xrow[kb2 + i];
                const _Float16 h = (_Float16)v;
                ah[8 + i] = h;
                al[8 + i] = (_Float16)(v - (float)h);
            }
            #pragma unroll
            for (int n = 0; n < 3; ++n) {
                acc[n] = __builtin_amdgcn_wmma_f32_16x16x32_f16(
                    false, ah, false, bh[n][ks], (short)0, acc[n], false, false);
                acc[n] = __builtin_amdgcn_wmma_f32_16x16x32_f16(
                    false, ah, false, bl[n][ks], (short)0, acc[n], false, false);
                acc[n] = __builtin_amdgcn_wmma_f32_16x16x32_f16(
                    false, al, false, bh[n][ks], (short)0, acc[n], false, false);
            }
        }

        // D layout: VGPR v -> M = v + 8*hi ; N = ln. Row-major [row][40] store.
        #pragma unroll
        for (int n = 0; n < 3; ++n) {
            const int j = n * 16 + ln;
            if (j < NJ) {
                #pragma unroll
                for (int v = 0; v < 8; ++v) {
                    const size_t m = (size_t)v + 8 * hi;
                    Zx[(rowBase + m) * NJ + j] = acc[n][v];
                }
            }
        }
    }
}

// =====================================================================
// Kernel B: sequential recurrence, one lane per batch row, one wave/block.
// Software-pipelined Zx loads (t+1 in flight during step t, t+2 prefetch).
// =====================================================================
__device__ __forceinline__ float fast_tanh(float x) {
#if __has_builtin(__builtin_amdgcn_tanhf)
    return __builtin_amdgcn_tanhf(x);
#else
    const float e = __expf(2.0f * x);
    return (e - 1.0f) / (e + 1.0f);
#endif
}
__device__ __forceinline__ float fast_sigmoid(float x) {
    return fmaf(0.5f, fast_tanh(0.5f * x), 0.5f);
}

__global__ void qlstm_recurrent(const float* __restrict__ Zx,
                                const float* __restrict__ Wf, const float* __restrict__ Wi,
                                const float* __restrict__ Wu, const float* __restrict__ Wo,
                                float* __restrict__ out) {
    // recurrent weights, transposed: whT[r][j] = W_g[j%10][64+r], j = g*10 + (j%10)
    __shared__ __align__(16) float whT[HH * NJ];
    for (int idx = threadIdx.x; idx < HH * NJ; idx += blockDim.x) {
        const int r = idx / NJ, j = idx % NJ;
        const int g = j / HH,  jr = j % HH;
        whT[idx] = gate_wptr(g, Wf, Wi, Wu, Wo)[(size_t)jr * DH + DD + r];
    }
    __syncthreads();

    const int b = blockIdx.x * blockDim.x + threadIdx.x;   // batch row
    if (b >= BB) return;

    float hx[HH], cx[HH];
    #pragma unroll
    for (int r = 0; r < HH; ++r) { hx[r] = 0.0f; cx[r] = 0.0f; }

    // preload step 0 row (row stride is 160B -> 16B aligned, b128 loads OK)
    float4 zbuf[NJ / 4];
    {
        const float4* z4 = (const float4*)(Zx + ((size_t)b) * NJ);
        #pragma unroll
        for (int q = 0; q < NJ / 4; ++q) zbuf[q] = z4[q];
    }

    for (int t = 0; t < TT; ++t) {
        // ---- unpack preloaded row, then immediately issue next row's loads ----
        float pre[NJ];
        #pragma unroll
        for (int q = 0; q < NJ / 4; ++q) {
            pre[4 * q + 0] = zbuf[q].x; pre[4 * q + 1] = zbuf[q].y;
            pre[4 * q + 2] = zbuf[q].z; pre[4 * q + 3] = zbuf[q].w;
        }
        {
            const int tn = (t + 1 < TT) ? (t + 1) : (TT - 1);
            const float4* z4 = (const float4*)(Zx + ((size_t)tn * BB + b) * NJ);
            #pragma unroll
            for (int q = 0; q < NJ / 4; ++q) zbuf[q] = z4[q];     // in flight during math
            const int tp = (t + 2 < TT) ? (t + 2) : (TT - 1);
            __builtin_prefetch(Zx + ((size_t)tp * BB + b) * NJ, 0, 3);
        }

        // ---- pre += hx @ Wh^T  (broadcast ds_read_b128 weights) ----
        #pragma unroll
        for (int r = 0; r < HH; ++r) {
            const float hr = hx[r];
            const float4* w4 = (const float4*)(whT + r * NJ);
            #pragma unroll
            for (int q = 0; q < NJ / 4; ++q) {
                const float4 w = w4[q];
                pre[4 * q + 0] = fmaf(hr, w.x, pre[4 * q + 0]);
                pre[4 * q + 1] = fmaf(hr, w.y, pre[4 * q + 1]);
                pre[4 * q + 2] = fmaf(hr, w.z, pre[4 * q + 2]);
                pre[4 * q + 3] = fmaf(hr, w.w, pre[4 * q + 3]);
            }
        }
        // ---- quantum gate: clip -> cos -> cumprod -> flip ----
        float gate[4][HH];
        #pragma unroll
        for (int g = 0; g < 4; ++g) {
            float cum = 1.0f;
            float cp[HH];
            #pragma unroll
            for (int r = 0; r < HH; ++r) {
                float p = fminf(fmaxf(pre[g * HH + r], -CLIPV), CLIPV);
                cum *= __cosf(p);
                cp[r] = cum;
            }
            #pragma unroll
            for (int r = 0; r < HH; ++r) gate[g][r] = cp[HH - 1 - r];
        }
        // ---- LSTM cell update ----
        #pragma unroll
        for (int r = 0; r < HH; ++r) {
            const float f = fast_sigmoid(gate[0][r]);
            const float i = fast_sigmoid(gate[1][r]);
            const float u = fast_tanh(gate[2][r]);
            const float o = fast_sigmoid(gate[3][r]);
            const float c = fmaf(f, cx[r], i * u);
            cx[r] = c;
            hx[r] = o * fast_tanh(c);
        }
        // ---- scalar modulation: hx *= cos(hx[9]) ----
        const float m = __cosf(hx[HH - 1]);
        #pragma unroll
        for (int r = 0; r < HH; ++r) hx[r] *= m;

        // ---- emit output [t][b][0..9] (40B row stride -> b64 stores) ----
        float2* op = (float2*)(out + ((size_t)t * BB + b) * HH);
        #pragma unroll
        for (int q = 0; q < HH / 2; ++q) op[q] = make_float2(hx[2 * q], hx[2 * q + 1]);
    }

    // ---- final (hx, cx) appended after outputs ----
    float2* hp = (float2*)(out + (size_t)TT * BB * HH + (size_t)b * HH);
    float2* cp = (float2*)(out + (size_t)TT * BB * HH + (size_t)BB * HH + (size_t)b * HH);
    #pragma unroll
    for (int q = 0; q < HH / 2; ++q) {
        hp[q] = make_float2(hx[2 * q], hx[2 * q + 1]);
        cp[q] = make_float2(cx[2 * q], cx[2 * q + 1]);
    }
}

// =====================================================================
extern "C" void kernel_launch(void* const* d_in, const int* in_sizes, int n_in,
                              void* d_out, int out_size, void* d_ws, size_t ws_size,
                              hipStream_t stream) {
    const float* x  = (const float*)d_in[0];
    const float* Wf = (const float*)d_in[1];
    const float* bf = (const float*)d_in[2];
    const float* Wi = (const float*)d_in[3];
    const float* bi = (const float*)d_in[4];
    const float* Wu = (const float*)d_in[5];
    const float* bu = (const float*)d_in[6];
    const float* Wo = (const float*)d_in[7];
    const float* bo = (const float*)d_in[8];
    float* out = (float*)d_out;
    float* Zx  = (float*)d_ws;          // needs T*B*40*4 = 160 MB scratch

    if (ws_size < (size_t)TT * BB * NJ * sizeof(float)) return;

    // Phase 1: bulk WMMA GEMM for the non-recurrent (input) part of all gates.
    qlstm_pregemm_wmma<<<2048, 128, 0, stream>>>(x, Wf, bf, Wi, bi, Wu, bu, Wo, bo, Zx);

    // Phase 2: sequential recurrence, one lane per batch row, one wave per block.
    qlstm_recurrent<<<BB / 32, 32, 0, stream>>>(Zx, Wf, Wi, Wu, Wo, out);
}
